// Self_AttentionLayer_52493090292085
// MI455X (gfx1250) — matrix operand order repaired and checked by reading, not compiled, not run on hardware
//
#include <hip/hip_runtime.h>

// ---------------------------------------------------------------------------
// Self-attention (SAGAN style) for MI455X / gfx1250, wave32 + WMMA bf16.
// B=4, C=512, H=W=64 -> N=4096, O=64.
// Pipeline:
//   1) convert x (f32 [B,C,N]) -> xT bf16 [B,N,C]   (proj B-operand layout)
//   2) fuse f_w/g_w/h_w -> Wc bf16 [640,512]
//   3) proj GEMM (WMMA 16x16x32 bf16, K=512): writes fT,gT bf16 [B,N,64],
//      h bf16 [B,C,N]  (+ biases)
//   4) flash-attention kernel, wave-specialized: 9 waves per block.
//      wave8 (producer): score tile S^T[nk,m] = sum_o f[o,nk] g[o,m] via WMMA,
//      online softmax over keys, packs exp(S) into WMMA-B layout bf16 into a
//      double-buffered LDS slot.  waves0-7 (consumers): 64 value channels each,
//      rescale + PV WMMA, overlapped with the producer's next block.
//      Epilogue: out = gamma * (acc / sum) + x  (f32).
// ---------------------------------------------------------------------------

typedef __attribute__((ext_vector_type(16))) __bf16     v16bf;
typedef __attribute__((ext_vector_type(8)))  float       v8f;
typedef __attribute__((ext_vector_type(4)))  unsigned int u32x4;

union BF16x16 {
    v16bf        bf;
    u32x4        q[2];
    unsigned int d[8];
};

#define WMMA_BF16(A, Bv, Cv) \
    __builtin_amdgcn_wmma_f32_16x16x32_bf16(false, (A), false, (Bv), (short)0, (Cv), false, false)

__device__ __forceinline__ unsigned short f2bf(float f) {
    unsigned int u = __float_as_uint(f);
    unsigned int r = u + 0x7FFFu + ((u >> 16) & 1u);
    return (unsigned short)(r >> 16);
}
__device__ __forceinline__ unsigned int pack2bf(float a, float b) {
    return (unsigned int)f2bf(a) | ((unsigned int)f2bf(b) << 16);
}

static constexpr int BB   = 4;
static constexpr int CC   = 512;
static constexpr int NN   = 4096;
static constexpr int OO   = 64;
static constexpr int ROWS = OO + OO + CC; // 640 fused projection rows

// ---------------------------------------------------------------------------
// Kernel 1: x [B,C,N] f32 -> xT [B,N,C] bf16
// ---------------------------------------------------------------------------
__global__ __launch_bounds__(256) void conv_x_kernel(const float* __restrict__ x,
                                                     unsigned short* __restrict__ xT) {
    const size_t total = (size_t)BB * CC * NN;
    for (size_t i = (size_t)blockIdx.x * blockDim.x + threadIdx.x; i < total;
         i += (size_t)gridDim.x * blockDim.x) {
        size_t b = i / ((size_t)CC * NN);
        size_t r = i - b * (size_t)CC * NN;
        size_t c = r / NN;
        size_t n = r - c * NN;
        xT[b * (size_t)NN * CC + n * CC + c] = f2bf(x[i]);
    }
}

// ---------------------------------------------------------------------------
// Kernel 2: fuse weights -> Wc [640,512] bf16  (rows: 0-63 f, 64-127 g, 128-639 h)
// ---------------------------------------------------------------------------
__global__ __launch_bounds__(256) void conv_w_kernel(const float* __restrict__ f_w,
                                                     const float* __restrict__ g_w,
                                                     const float* __restrict__ h_w,
                                                     unsigned short* __restrict__ Wc) {
    const int total = ROWS * CC;
    for (int i = blockIdx.x * blockDim.x + threadIdx.x; i < total;
         i += gridDim.x * blockDim.x) {
        int row = i / CC, c = i - row * CC;
        float v;
        if (row < OO)            v = f_w[row * CC + c];
        else if (row < 2 * OO)   v = g_w[(row - OO) * CC + c];
        else                     v = h_w[(row - 2 * OO) * CC + c];
        Wc[i] = f2bf(v);
    }
}

// ---------------------------------------------------------------------------
// Kernel 3: projection GEMM  P[row,n] = sum_c Wc[row,c] * x[b,c,n] + bias
// grid = (N/128, ROWS/16, B), block = 256 (8 waves, 1 output tile per wave)
// ---------------------------------------------------------------------------
__global__ __launch_bounds__(256) void proj_kernel(const unsigned short* __restrict__ Wc,
                                                   const unsigned short* __restrict__ xT,
                                                   const float* __restrict__ f_b,
                                                   const float* __restrict__ g_b,
                                                   const float* __restrict__ h_b,
                                                   unsigned short* __restrict__ fT,
                                                   unsigned short* __restrict__ gT,
                                                   unsigned short* __restrict__ hB) {
    const int wv = threadIdx.x >> 5;
    const int lane = threadIdx.x & 31;
    const int hi = lane >> 4, lm = lane & 15;
    const int b  = blockIdx.z;
    const int m0 = blockIdx.y * 16;                  // output-row tile
    const int n0 = (blockIdx.x * 8 + wv) * 16;       // pixel tile

    const unsigned short* xTb = xT + (size_t)b * NN * CC;

    v8f acc = {0.f, 0.f, 0.f, 0.f, 0.f, 0.f, 0.f, 0.f};
    const size_t arow = (size_t)(m0 + lm) * CC;
    const size_t brow = (size_t)(n0 + lm) * CC;

#pragma unroll
    for (int cs = 0; cs < CC; cs += 32) {
        BF16x16 A, Bv;
        // A 16x32 bf16 layout: lane row = lm, K runs {8*hi + i, 8*hi + 16 + i}
        A.q[0] = *(const u32x4*)(Wc + arow + cs + 8 * hi);
        A.q[1] = *(const u32x4*)(Wc + arow + cs + 8 * hi + 16);
        // B 32x16 bf16 layout: lane col = lm, K runs {8*hi + i, 8*hi + 16 + i}
        Bv.q[0] = *(const u32x4*)(xTb + brow + cs + 8 * hi);
        Bv.q[1] = *(const u32x4*)(xTb + brow + cs + 8 * hi + 16);
        acc = WMMA_BF16(A.bf, Bv.bf, acc);
    }

    // D layout: col n = n0+lm, row = m0 + r + 8*hi
    unsigned short* fTb = fT + (size_t)b * NN * OO;
    unsigned short* gTb = gT + (size_t)b * NN * OO;
    unsigned short* hBb = hB + (size_t)b * CC * NN;
    const int n = n0 + lm;
#pragma unroll
    for (int r = 0; r < 8; ++r) {
        int row = m0 + r + 8 * hi;
        float bias = (row < OO) ? f_b[row]
                   : (row < 2 * OO) ? g_b[row - OO] : h_b[row - 2 * OO];
        float v = acc[r] + bias;
        unsigned short bf = f2bf(v);
        if (row < OO)            fTb[(size_t)n * OO + row] = bf;
        else if (row < 2 * OO)   gTb[(size_t)n * OO + (row - OO)] = bf;
        else                     hBb[(size_t)(row - 2 * OO) * NN + n] = bf;
    }
}

// ---------------------------------------------------------------------------
// Kernel 4: flash attention, wave-specialized producer/consumer.
// grid = (N/16, B), block = 288 (9 waves).  Query tile of 16 per block.
// wave8 produces exp(S) tiles (WMMA-B layout) + corr into double-buffered LDS;
// waves0-7 each own value channels [64w, 64w+64) and run the PV WMMAs.
// One __syncthreads per 32-key block; producer of block i+1 overlaps
// consumers of block i.
// ---------------------------------------------------------------------------
__global__ __launch_bounds__(288) void attn_kernel(const unsigned short* __restrict__ fT,
                                                   const unsigned short* __restrict__ gT,
                                                   const unsigned short* __restrict__ hB,
                                                   const float* __restrict__ x,
                                                   const float* __restrict__ gamma_p,
                                                   float* __restrict__ out) {
    __shared__ u32x4 sP[2][32][2];   // double-buffered exp(S) tile, WMMA-B bf16 layout
    __shared__ float sCorr[2][16];   // per-query rescale factor
    __shared__ float sInv[16];       // per-query 1/sum

    const int wv   = threadIdx.x >> 5;
    const int lane = threadIdx.x & 31;
    const int hi = lane >> 4, lm = lane & 15;
    const int b  = blockIdx.y;
    const int m0 = blockIdx.x * 16;
    const int m  = m0 + lm;                      // this lane's query column
    const int ITERS = NN / 32;

    if (wv == 8) {
        // ---------------- producer wave: scores + online softmax ----------------
        const unsigned short* fTb = fT + (size_t)b * NN * OO;
        const unsigned short* gTb = gT + (size_t)b * NN * OO;

        // Q operand (B role), loop-invariant: B[k=o][col=m] = g[o,m] = gT[m][o]
        BF16x16 Bg0, Bg1;
        {
            const size_t gr = (size_t)m * OO;
            Bg0.q[0] = *(const u32x4*)(gTb + gr + 8 * hi);
            Bg0.q[1] = *(const u32x4*)(gTb + gr + 8 * hi + 16);
            Bg1.q[0] = *(const u32x4*)(gTb + gr + 32 + 8 * hi);
            Bg1.q[1] = *(const u32x4*)(gTb + gr + 32 + 8 * hi + 16);
        }

        float run_max = -3.0e38f, run_sum = 0.f;

        for (int it = 0; it < ITERS; ++it) {
            const int nk0 = it * 32;
            // S^T tiles: rows nk (keys), cols m (queries).  A[nk][o]=f[o,nk]=fT[nk][o]
            v8f S0 = (v8f){0.f,0.f,0.f,0.f,0.f,0.f,0.f,0.f};
            v8f S1 = (v8f){0.f,0.f,0.f,0.f,0.f,0.f,0.f,0.f};
            BF16x16 Af;
            const size_t fr0 = (size_t)(nk0 + lm) * OO;
            __builtin_prefetch(fTb + fr0 + 32 * OO, 0, 0);   // next key block
            Af.q[0] = *(const u32x4*)(fTb + fr0 + 8 * hi);
            Af.q[1] = *(const u32x4*)(fTb + fr0 + 8 * hi + 16);
            S0 = WMMA_BF16(Af.bf, Bg0.bf, S0);
            Af.q[0] = *(const u32x4*)(fTb + fr0 + 32 + 8 * hi);
            Af.q[1] = *(const u32x4*)(fTb + fr0 + 32 + 8 * hi + 16);
            S0 = WMMA_BF16(Af.bf, Bg1.bf, S0);
            const size_t fr1 = (size_t)(nk0 + 16 + lm) * OO;
            Af.q[0] = *(const u32x4*)(fTb + fr1 + 8 * hi);
            Af.q[1] = *(const u32x4*)(fTb + fr1 + 8 * hi + 16);
            S1 = WMMA_BF16(Af.bf, Bg0.bf, S1);
            Af.q[0] = *(const u32x4*)(fTb + fr1 + 32 + 8 * hi);
            Af.q[1] = *(const u32x4*)(fTb + fr1 + 32 + 8 * hi + 16);
            S1 = WMMA_BF16(Af.bf, Bg1.bf, S1);

            // Online softmax over keys for this lane's query m.
            float tm = S0[0];
#pragma unroll
            for (int r = 0; r < 8; ++r) { tm = fmaxf(tm, S0[r]); tm = fmaxf(tm, S1[r]); }
            tm = fmaxf(tm, __shfl_xor(tm, 16, 32));
            float nmax = fmaxf(run_max, tm);
            float corr = __expf(run_max - nmax);
            run_max = nmax;

            float e0[8], e1[8], lsum = 0.f;
#pragma unroll
            for (int r = 0; r < 8; ++r) {
                e0[r] = __expf(S0[r] - nmax);
                e1[r] = __expf(S1[r] - nmax);
                lsum += e0[r] + e1[r];
            }
            lsum += __shfl_xor(lsum, 16, 32);
            run_sum = run_sum * corr + lsum;

            // Re-layout D -> B operand: lane needs K = 16*hi + i  (keys).
            // hi=0 lanes: K 0..7 = own S0 regs, K 8..15 = partner's S0 regs.
            // hi=1 lanes: K16..23 = partner's S1 regs, K24..31 = own S1 regs.
            float p0[8], p1[8];
#pragma unroll
            for (int r = 0; r < 8; ++r) {
                p0[r] = __shfl_xor(e0[r], 16, 32);
                p1[r] = __shfl_xor(e1[r], 16, 32);
            }
            BF16x16 P;
#pragma unroll
            for (int j = 0; j < 4; ++j) {
                float a0 = hi ? p1[2 * j]     : e0[2 * j];
                float a1 = hi ? p1[2 * j + 1] : e0[2 * j + 1];
                float b0 = hi ? e1[2 * j]     : p0[2 * j];
                float b1 = hi ? e1[2 * j + 1] : p0[2 * j + 1];
                P.d[j]     = pack2bf(a0, a1);
                P.d[4 + j] = pack2bf(b0, b1);
            }
            const int buf = it & 1;
            sP[buf][lane][0] = P.q[0];
            sP[buf][lane][1] = P.q[1];
            sCorr[buf][lm] = corr;
            __syncthreads();                       // publish block `it`
        }
        sInv[lm] = 1.0f / run_sum;                 // lane pair writes same value
        __syncthreads();                           // final publish
        // producer wave exits; no output duty
    } else {
        // ---------------- consumer waves: PV accumulation ----------------
        const unsigned short* hBb = hB + (size_t)b * CC * NN;
        const int cbase = wv * 64;

        v8f acc[4];
#pragma unroll
        for (int t = 0; t < 4; ++t) acc[t] = (v8f){0.f,0.f,0.f,0.f,0.f,0.f,0.f,0.f};

        for (int it = 0; it < ITERS; ++it) {
            __syncthreads();                       // wait for block `it`
            const int nk0 = it * 32;
            const int buf = it & 1;
            const float cr = sCorr[buf][lm];
            BF16x16 Bp;
            Bp.q[0] = sP[buf][lane][0];
            Bp.q[1] = sP[buf][lane][1];
#pragma unroll
            for (int t = 0; t < 4; ++t) {
                acc[t] = acc[t] * cr;
                BF16x16 Ah;   // A[c][k=n] = h[c, nk0+k]; K = 16hi+i (contiguous)
                const size_t hr = (size_t)(cbase + t * 16 + lm) * NN + nk0;
                __builtin_prefetch(hBb + hr + 32, 0, 0);   // next key block
                Ah.q[0] = *(const u32x4*)(hBb + hr + 16 * hi);
                Ah.q[1] = *(const u32x4*)(hBb + hr + 16 * hi + 8);
                acc[t] = WMMA_BF16(Ah.bf, Bp.bf, acc[t]);
            }
        }
        __syncthreads();                           // matches producer's final sync

        const float inv   = sInv[lm];
        const float gamma = gamma_p[0];
        const float* xb = x   + (size_t)b * CC * NN;
        float*       ob = out + (size_t)b * CC * NN;
#pragma unroll
        for (int t = 0; t < 4; ++t) {
#pragma unroll
            for (int r = 0; r < 8; ++r) {
                int c = cbase + t * 16 + r + 8 * hi;
                size_t idx = (size_t)c * NN + m;
                ob[idx] = gamma * (acc[t][r] * inv) + xb[idx];
            }
        }
    }
}

// ---------------------------------------------------------------------------
// Host launcher
// ---------------------------------------------------------------------------
extern "C" void kernel_launch(void* const* d_in, const int* in_sizes, int n_in,
                              void* d_out, int out_size, void* d_ws, size_t ws_size,
                              hipStream_t stream) {
    const float* x     = (const float*)d_in[0];
    const float* f_w   = (const float*)d_in[1];
    const float* f_b   = (const float*)d_in[2];
    const float* g_w   = (const float*)d_in[3];
    const float* g_b   = (const float*)d_in[4];
    const float* h_w   = (const float*)d_in[5];
    const float* h_b   = (const float*)d_in[6];
    const float* gamma = (const float*)d_in[7];
    float* out = (float*)d_out;

    // Workspace layout (bf16 = unsigned short), all region sizes 256B aligned.
    char* ws = (char*)d_ws;
    size_t off = 0;
    unsigned short* xT = (unsigned short*)(ws + off); off += (size_t)BB * NN * CC * 2;   // 16 MB
    unsigned short* Wc = (unsigned short*)(ws + off); off += (size_t)ROWS * CC * 2;      // 640 KB
    unsigned short* fT = (unsigned short*)(ws + off); off += (size_t)BB * NN * OO * 2;   // 2 MB
    unsigned short* gT = (unsigned short*)(ws + off); off += (size_t)BB * NN * OO * 2;   // 2 MB
    unsigned short* hB = (unsigned short*)(ws + off); off += (size_t)BB * CC * NN * 2;   // 16 MB
    (void)off; (void)ws_size; (void)in_sizes; (void)n_in; (void)out_size;

    conv_x_kernel<<<4096, 256, 0, stream>>>(x, xT);
    conv_w_kernel<<<512, 256, 0, stream>>>(f_w, g_w, h_w, Wc);

    dim3 pgrid(NN / 128, ROWS / 16, BB);   // (32, 40, 4)
    proj_kernel<<<pgrid, 256, 0, stream>>>(Wc, xT, f_b, g_b, h_b, fT, gT, hB);

    dim3 agrid(NN / 16, BB);               // (256, 4)
    attn_kernel<<<agrid, 288, 0, stream>>>(fT, gT, hB, x, gamma, out);
}
